// Attn_65034394796889
// MI455X (gfx1250) — matrix operand order repaired and checked by reading, not compiled, not run on hardware
//
#include <hip/hip_runtime.h>
#include <hip/hip_bf16.h>

// ---------------------------------------------------------------------------
// MI455X (gfx1250) GQA attention layer.
//   K1/K3 GEMMs: 4-wave workgroups, 128x64 tile; B tile (64x32 bf16) staged in
//   LDS by the Tensor Data Mover (double-buffered, overlapped with WMMA).
//   K2: sliding-window flash attention, one wave per (head, 16-query block).
// All matmuls: v_wmma_f32_16x16x32_bf16 (bf16 in, f32 accumulate). Wave32.
// ---------------------------------------------------------------------------

typedef __attribute__((ext_vector_type(16))) __bf16 v16bf;
typedef __attribute__((ext_vector_type(8)))  __bf16 v8bf;
typedef __attribute__((ext_vector_type(8)))  float  v8f;
typedef __attribute__((ext_vector_type(4)))  unsigned int v4u;
typedef __attribute__((ext_vector_type(8)))  int v8i;
typedef __attribute__((ext_vector_type(4)))  int v4i;

#define SEQ     2048
#define DMODEL  2048
#define NHEADS  32
#define NKV     8
#define HD      64
#define QSIZE   (NHEADS*HD)          // 2048
#define KVSIZE  (NKV*HD)             // 512
#define OTOT    (QSIZE + 2*KVSIZE)   // 3072
#define WINDOW  512
#define LOG2E   1.4426950408889634f
#define NEGBIG  (-1.0e30f)

#if __has_builtin(__builtin_amdgcn_tensor_load_to_lds) && \
    __has_builtin(__builtin_amdgcn_s_wait_tensorcnt)
#define HAVE_TDM 1
#else
#define HAVE_TDM 0
#endif

__device__ __forceinline__ v8f wmma_bf16(v16bf a, v16bf b, v8f c) {
    return __builtin_amdgcn_wmma_f32_16x16x32_bf16(
        false, a, false, b, (short)0, c, false, false);
}

// A-fragment: 16x32 bf16 row-major. lanes 0-15 -> row=lane, K={0..7,16..23};
// lanes 16-31 -> row=lane-16, K={8..15,24..31}.
__device__ __forceinline__ v16bf load_frag_a(const __bf16* base, int ld, int lane) {
    const int row  = lane & 15;
    const int koff = (lane >> 4) << 3;
    const __bf16* p = base + (size_t)row * ld + koff;
    union { v16bf v; v8bf h[2]; } u;
    u.h[0] = *(const v8bf*)(p);
    u.h[1] = *(const v8bf*)(p + 16);
    return u.v;
}

// B-fragment: 32x16 (KxN), column n contiguous along K at base + n*ld.
// lanes 0-15 -> col=lane, K={0..7,16..23}; lanes 16-31 -> col=lane-16, K={8..15,24..31}.
__device__ __forceinline__ v16bf load_frag_b(const __bf16* base, int ld, int lane) {
    const int col = lane & 15;
    const int kb  = (lane >> 4) << 3;
    const __bf16* p = base + (size_t)col * ld + kb;
    union { v16bf v; v8bf h[2]; } u;
    u.h[0] = *(const v8bf*)(p);
    u.h[1] = *(const v8bf*)(p + 16);
    return u.v;
}

// ---------------------------------------------------------------------------
// TDM: stage a rows x 32 bf16 tile (row pitch = ld elements) into LDS,
// packed contiguously as [row][32]. 2D descriptor per ISA 08_async_tensor §8.
// This toolchain exposes the 6-arg builtin:
//   (v4u g0, v8i g1, v4i g2, v4i g3, v8i g4, i32 cpol)
// Groups 2/3 are only needed for >2D tensors -> zero-filled here.
// ---------------------------------------------------------------------------
__device__ __forceinline__ void tdm_load_b_tile(const __bf16* gsrc, unsigned lds_off,
                                                unsigned rows, unsigned ld,
                                                unsigned tensor_rows) {
#if HAVE_TDM
    const unsigned long long ga = (unsigned long long)(uintptr_t)gsrc;
    v4u g0;
    g0[0] = 1u;                                        // count=1, user mode
    g0[1] = lds_off;                                   // lds_addr (bytes)
    g0[2] = (unsigned)ga;                              // global_addr[31:0]
    g0[3] = (unsigned)((ga >> 32) & 0x01FFFFFFull) | (2u << 30); // addr[56:32], type=2
    const unsigned td0 = ld;                           // tensor_dim0 (elements)
    const unsigned td1 = tensor_rows;                  // tensor_dim1
    v8i g1;
    g1[0] = (int)(1u << 16);                           // data_size=1 (2 bytes)
    g1[1] = (int)((td0 & 0xffffu) << 16);              // tensor_dim0[15:0]
    g1[2] = (int)((td0 >> 16) | ((td1 & 0xffffu) << 16));
    g1[3] = (int)((td1 >> 16) | (32u << 16));          // tile_dim0 = 32
    g1[4] = (int)rows;                                 // tile_dim1 = rows, tile_dim2=0
    g1[5] = (int)ld;                                   // tensor_dim0_stride[31:0]
    g1[6] = 0;                                         // stride hi / dim1_stride lo
    g1[7] = 0;
    v4i gz4 = {0, 0, 0, 0};
    v8i gz8 = {0, 0, 0, 0, 0, 0, 0, 0};
    __builtin_amdgcn_tensor_load_to_lds(g0, g1, gz4, gz4, gz8, 0);
#else
    (void)gsrc; (void)lds_off; (void)rows; (void)ld; (void)tensor_rows;
#endif
}

__device__ __forceinline__ void tdm_wait0() {
#if HAVE_TDM
    __builtin_amdgcn_s_wait_tensorcnt(0);
#endif
}

// Fallback cooperative copy (used only if TDM builtin is unavailable).
__device__ __forceinline__ void coop_copy_b_tile(const __bf16* gsrc, __bf16* dst,
                                                 unsigned rows, unsigned ld, int tid,
                                                 int nthreads) {
#if !HAVE_TDM
    for (unsigned r = tid; r < rows; r += nthreads) {
        v8bf a = *(const v8bf*)(gsrc + (size_t)r * ld);
        v8bf b = *(const v8bf*)(gsrc + (size_t)r * ld + 8);
        v8bf c = *(const v8bf*)(gsrc + (size_t)r * ld + 16);
        v8bf d = *(const v8bf*)(gsrc + (size_t)r * ld + 24);
        *(v8bf*)(dst + r * 32)      = a;
        *(v8bf*)(dst + r * 32 + 8)  = b;
        *(v8bf*)(dst + r * 32 + 16) = c;
        *(v8bf*)(dst + r * 32 + 24) = d;
    }
#else
    (void)gsrc; (void)dst; (void)rows; (void)ld; (void)tid; (void)nthreads;
#endif
}

// ---------------------------------------------------------------------------
// K0: f32 -> bf16 convert
// ---------------------------------------------------------------------------
__global__ void cvt_f32_to_bf16(const float* __restrict__ in,
                                __bf16* __restrict__ out, int n) {
    int i = blockIdx.x * blockDim.x + threadIdx.x;
    if (i < n) out[i] = (__bf16)in[i];
}

// ---------------------------------------------------------------------------
// K1: qkv = x @ w_qkv^T with fused RMSNorm+RoPE epilogue.
// 4 waves/block; block tile = 128 rows x 64-col head chunk. B via TDM->LDS.
// ---------------------------------------------------------------------------
__global__ void __launch_bounds__(128)
qkv_gemm_norm_rope(const __bf16* __restrict__ xb,
                   const __bf16* __restrict__ wb,
                   const float*  __restrict__ qnw,
                   const float*  __restrict__ knw,
                   __bf16* __restrict__ qout,   // [32][2048][64]
                   __bf16* __restrict__ kout,   // [8][2048][64]
                   __bf16* __restrict__ vTout)  // [8][64][2048]
{
    __shared__ __align__(16) __bf16 bsh[2][64 * 32];
    const int tid  = threadIdx.x;
    const int lane = tid & 31;
    const int w    = tid >> 5;
    const int m0   = blockIdx.x * 128 + w * 32;
    const int h    = blockIdx.y;         // 0..47
    const int o0   = h * 64;
    const bool hi  = lane >= 16;
    const int colq = lane & 15;

    if (w == 0)
        tdm_load_b_tile(wb + (size_t)o0 * DMODEL, (unsigned)(uintptr_t)&bsh[0][0],
                        64, DMODEL, OTOT);
    coop_copy_b_tile(wb + (size_t)o0 * DMODEL, &bsh[0][0], 64, DMODEL, tid, 128);

    v8f acc[2][4] = {};
    for (int kk = 0; kk < DMODEL; kk += 32) {
        const int cur = (kk >> 5) & 1;
        if (w == 0) tdm_wait0();
        __syncthreads();
        if (kk + 32 < DMODEL) {
            if (w == 0)
                tdm_load_b_tile(wb + (size_t)o0 * DMODEL + kk + 32,
                                (unsigned)(uintptr_t)&bsh[cur ^ 1][0], 64, DMODEL, OTOT);
            coop_copy_b_tile(wb + (size_t)o0 * DMODEL + kk + 32, &bsh[cur ^ 1][0],
                             64, DMODEL, tid, 128);
        }
        __builtin_prefetch(xb + (size_t)m0 * DMODEL + kk + 256, 0, 1);
        v16bf a0 = load_frag_a(xb + (size_t)m0        * DMODEL + kk, DMODEL, lane);
        v16bf a1 = load_frag_a(xb + (size_t)(m0 + 16) * DMODEL + kk, DMODEL, lane);
#pragma unroll
        for (int nt = 0; nt < 4; ++nt) {
            v16bf b = load_frag_b(&bsh[cur][nt * 16 * 32], 32, lane);
            acc[0][nt] = wmma_bf16(a0, b, acc[0][nt]);
            acc[1][nt] = wmma_bf16(a1, b, acc[1][nt]);
        }
        __syncthreads();
    }

    if (h >= 40) {                       // ---- V head: convert + transposed store
        const int vh = h - 40;
#pragma unroll
        for (int rh = 0; rh < 2; ++rh) {
            const int s0 = m0 + rh * 16 + (hi ? 8 : 0);
#pragma unroll
            for (int nt = 0; nt < 4; ++nt) {
                const int d = nt * 16 + colq;
                v8bf pk;
#pragma unroll
                for (int r = 0; r < 8; ++r) pk[r] = (__bf16)acc[rh][nt][r];
                *(v8bf*)(vTout + ((size_t)(vh * 64 + d)) * SEQ + s0) = pk;
            }
        }
        return;
    }

    // ---- Q or K head: RMSNorm + RoPE
    const bool isq = (h < 32);
    const float* nw = isq ? qnw : knw;
    float wcol[4];
#pragma unroll
    for (int nt = 0; nt < 4; ++nt) wcol[nt] = nw[nt * 16 + colq];
    const float inf0 = exp2f(-0.41524101186092029f * (float)colq);        // 10000^(-i/32)
    const float inf1 = exp2f(-0.41524101186092029f * (float)(16 + colq));
    const float qscale = isq ? (0.125f * LOG2E) : 1.0f;  // fold softmax scale*log2e into Q
    __bf16* outp = isq ? (qout + (size_t)h * SEQ * HD)
                       : (kout + (size_t)(h - 32) * SEQ * HD);

#pragma unroll
    for (int rh = 0; rh < 2; ++rh) {
#pragma unroll
        for (int r = 0; r < 8; ++r) {
            float s2 = 0.f;
#pragma unroll
            for (int nt = 0; nt < 4; ++nt) s2 += acc[rh][nt][r] * acc[rh][nt][r];
            s2 += __shfl_xor(s2, 1, 32);
            s2 += __shfl_xor(s2, 2, 32);
            s2 += __shfl_xor(s2, 4, 32);
            s2 += __shfl_xor(s2, 8, 32);
            const float rms = rsqrtf(s2 * (1.0f / 64.0f) + 1e-6f);
            const int srow = m0 + rh * 16 + r + (hi ? 8 : 0);
            const float n0 = acc[rh][0][r] * rms * wcol[0];
            const float n1 = acc[rh][1][r] * rms * wcol[1];
            const float n2 = acc[rh][2][r] * rms * wcol[2];
            const float n3 = acc[rh][3][r] * rms * wcol[3];
            const float a0 = (float)srow * inf0, a1 = (float)srow * inf1;
            const float c0 = cosf(a0), s0 = sinf(a0);
            const float c1 = cosf(a1), s1 = sinf(a1);
            __bf16* row = outp + (size_t)srow * HD;
            row[colq]      = (__bf16)((n0 * c0 - n2 * s0) * qscale);
            row[16 + colq] = (__bf16)((n1 * c1 - n3 * s1) * qscale);
            row[32 + colq] = (__bf16)((n2 * c0 + n0 * s0) * qscale);
            row[48 + colq] = (__bf16)((n3 * c1 + n1 * s1) * qscale);
        }
    }
}

// ---------------------------------------------------------------------------
// K2: sliding-window flash attention. One wave per (head, 16-query block).
// Online softmax in exp2 domain (scale*log2e pre-folded into Q).
// ---------------------------------------------------------------------------
__global__ void flash_attn(const __bf16* __restrict__ qg,
                           const __bf16* __restrict__ kg,
                           const __bf16* __restrict__ vTg,
                           __bf16* __restrict__ attn_out) // [2048][2048]
{
    __shared__ __align__(16) __bf16 plds[16 * 32];
    const int lane = threadIdx.x & 31;
    const int q0   = blockIdx.x * 16;
    const int h    = blockIdx.y;
    const int kvh  = h >> 2;
    const bool hi  = lane >= 16;
    const int colq = lane & 15;

    const v16bf qA0 = load_frag_a(qg + ((size_t)h * SEQ + q0) * HD + 0,  HD, lane);
    const v16bf qA1 = load_frag_a(qg + ((size_t)h * SEQ + q0) * HD + 32, HD, lane);

    v8f o[4] = {};
    float m[8], l[8];
#pragma unroll
    for (int r = 0; r < 8; ++r) { m[r] = NEGBIG; l[r] = 0.f; }

    int kstart = q0 - (WINDOW - 1);
    if (kstart < 0) kstart = 0;
    kstart &= ~31;
    const int kend = q0 + 15;

    for (int kb = kstart; kb <= kend; kb += 32) {
        v8f S[2];
#pragma unroll
        for (int sub = 0; sub < 2; ++sub) {
            const int k16 = kb + sub * 16;
            v16bf kB0 = load_frag_b(kg + ((size_t)kvh * SEQ + k16) * HD + 0,  HD, lane);
            v16bf kB1 = load_frag_b(kg + ((size_t)kvh * SEQ + k16) * HD + 32, HD, lane);
            v8f s = {};
            s = wmma_bf16(qA0, kB0, s);
            s = wmma_bf16(qA1, kB1, s);
            const int kidx = k16 + colq;
#pragma unroll
            for (int r = 0; r < 8; ++r) {
                const int qrow = q0 + r + (hi ? 8 : 0);
                const bool keep = (kidx <= qrow) && ((qrow - kidx) < WINDOW);
                s[r] = keep ? s[r] : NEGBIG;
            }
            S[sub] = s;
        }
#pragma unroll
        for (int r = 0; r < 8; ++r) {
            float t = fmaxf(S[0][r], S[1][r]);
            t = fmaxf(t, __shfl_xor(t, 1, 32));
            t = fmaxf(t, __shfl_xor(t, 2, 32));
            t = fmaxf(t, __shfl_xor(t, 4, 32));
            t = fmaxf(t, __shfl_xor(t, 8, 32));
            const float mn    = fmaxf(m[r], t);
            const float alpha = exp2f(m[r] - mn);
            const float p0    = exp2f(S[0][r] - mn);
            const float p1    = exp2f(S[1][r] - mn);
            float rs = p0 + p1;
            rs += __shfl_xor(rs, 1, 32);
            rs += __shfl_xor(rs, 2, 32);
            rs += __shfl_xor(rs, 4, 32);
            rs += __shfl_xor(rs, 8, 32);
            l[r] = l[r] * alpha + rs;
            m[r] = mn;
#pragma unroll
            for (int nt = 0; nt < 4; ++nt) o[nt][r] = o[nt][r] * alpha;
            const int rl = r + (hi ? 8 : 0);
            plds[rl * 32 + colq]      = (__bf16)p0;
            plds[rl * 32 + 16 + colq] = (__bf16)p1;
        }
        __syncthreads();
        const v16bf pA = load_frag_a(plds, 32, lane);
        __syncthreads();
#pragma unroll
        for (int nt = 0; nt < 4; ++nt) {
            v16bf vB = load_frag_b(vTg + (size_t)(kvh * 64 + nt * 16) * SEQ + kb, SEQ, lane);
            o[nt] = wmma_bf16(pA, vB, o[nt]);
        }
    }

#pragma unroll
    for (int r = 0; r < 8; ++r) {
        const float inv = 1.0f / l[r];
        const int srow = q0 + r + (hi ? 8 : 0);
        __bf16* row = attn_out + (size_t)srow * QSIZE + h * HD;
#pragma unroll
        for (int nt = 0; nt < 4; ++nt)
            row[nt * 16 + colq] = (__bf16)(o[nt][r] * inv);
    }
}

// ---------------------------------------------------------------------------
// K3: out = attn @ w_out^T (f32 to d_out). Same TDM-staged structure as K1.
// ---------------------------------------------------------------------------
__global__ void __launch_bounds__(128)
out_gemm(const __bf16* __restrict__ ab,
         const __bf16* __restrict__ wb,
         float* __restrict__ out)
{
    __shared__ __align__(16) __bf16 bsh[2][64 * 32];
    const int tid  = threadIdx.x;
    const int lane = tid & 31;
    const int w    = tid >> 5;
    const int m0 = blockIdx.x * 128 + w * 32;
    const int n0 = blockIdx.y * 64;
    const bool hi = lane >= 16;
    const int colq = lane & 15;

    if (w == 0)
        tdm_load_b_tile(wb + (size_t)n0 * QSIZE, (unsigned)(uintptr_t)&bsh[0][0],
                        64, QSIZE, DMODEL);
    coop_copy_b_tile(wb + (size_t)n0 * QSIZE, &bsh[0][0], 64, QSIZE, tid, 128);

    v8f acc[2][4] = {};
    for (int kk = 0; kk < QSIZE; kk += 32) {
        const int cur = (kk >> 5) & 1;
        if (w == 0) tdm_wait0();
        __syncthreads();
        if (kk + 32 < QSIZE) {
            if (w == 0)
                tdm_load_b_tile(wb + (size_t)n0 * QSIZE + kk + 32,
                                (unsigned)(uintptr_t)&bsh[cur ^ 1][0], 64, QSIZE, DMODEL);
            coop_copy_b_tile(wb + (size_t)n0 * QSIZE + kk + 32, &bsh[cur ^ 1][0],
                             64, QSIZE, tid, 128);
        }
        __builtin_prefetch(ab + (size_t)m0 * QSIZE + kk + 256, 0, 1);
        v16bf a0 = load_frag_a(ab + (size_t)m0        * QSIZE + kk, QSIZE, lane);
        v16bf a1 = load_frag_a(ab + (size_t)(m0 + 16) * QSIZE + kk, QSIZE, lane);
#pragma unroll
        for (int nt = 0; nt < 4; ++nt) {
            v16bf b = load_frag_b(&bsh[cur][nt * 16 * 32], 32, lane);
            acc[0][nt] = wmma_bf16(a0, b, acc[0][nt]);
            acc[1][nt] = wmma_bf16(a1, b, acc[1][nt]);
        }
        __syncthreads();
    }
#pragma unroll
    for (int rh = 0; rh < 2; ++rh)
#pragma unroll
        for (int r = 0; r < 8; ++r) {
            const int srow = m0 + rh * 16 + r + (hi ? 8 : 0);
#pragma unroll
            for (int nt = 0; nt < 4; ++nt)
                out[(size_t)srow * DMODEL + n0 + nt * 16 + colq] = acc[rh][nt][r];
        }
}

// ---------------------------------------------------------------------------
// Launch
// ---------------------------------------------------------------------------
extern "C" void kernel_launch(void* const* d_in, const int* in_sizes, int n_in,
                              void* d_out, int out_size, void* d_ws, size_t ws_size,
                              hipStream_t stream) {
    const float* x     = (const float*)d_in[0];
    const float* w_qkv = (const float*)d_in[1];
    const float* w_out = (const float*)d_in[2];
    const float* qnw   = (const float*)d_in[3];
    const float* knw   = (const float*)d_in[4];
    float* out = (float*)d_out;

    char* ws = (char*)d_ws;                                        // ~50.3 MB total
    __bf16* x_bf    = (__bf16*)(ws);                               // 8 MB
    __bf16* wqkv_bf = (__bf16*)(ws + 8388608);                     // 12.6 MB
    __bf16* wout_bf = (__bf16*)(ws + 20971520);                    // 8 MB
    __bf16* q_bf    = (__bf16*)(ws + 29360128);                    // 8 MB
    __bf16* k_bf    = (__bf16*)(ws + 37748736);                    // 2 MB
    __bf16* vT_bf   = (__bf16*)(ws + 39845888);                    // 2 MB
    __bf16* attn_bf = (__bf16*)(ws + 41943040);                    // 8 MB

    const int nx = SEQ * DMODEL;
    const int nq = OTOT * DMODEL;
    cvt_f32_to_bf16<<<(nx + 255) / 256, 256, 0, stream>>>(x,     x_bf,    nx);
    cvt_f32_to_bf16<<<(nq + 255) / 256, 256, 0, stream>>>(w_qkv, wqkv_bf, nq);
    cvt_f32_to_bf16<<<(nx + 255) / 256, 256, 0, stream>>>(w_out, wout_bf, nx);

    qkv_gemm_norm_rope<<<dim3(SEQ / 128, OTOT / 64), 128, 0, stream>>>(
        x_bf, wqkv_bf, qnw, knw, q_bf, k_bf, vT_bf);

    flash_attn<<<dim3(SEQ / 16, NHEADS), 32, 0, stream>>>(q_bf, k_bf, vT_bf, attn_bf);

    out_gemm<<<dim3(SEQ / 128, DMODEL / 64), 128, 0, stream>>>(attn_bf, wout_bf, out);
}
